// SpikingMambaBlock_41120016892141
// MI455X (gfx1250) — matrix operand order repaired and checked by reading, not compiled, not run on hardware
//
#include <hip/hip_runtime.h>

typedef unsigned short u16;
typedef __attribute__((ext_vector_type(16))) __bf16 v16bf;
typedef __attribute__((ext_vector_type(8)))  float  v8f;
typedef __attribute__((ext_vector_type(4))) unsigned int u32x4;
typedef __attribute__((ext_vector_type(8))) int i32x8;
typedef __attribute__((ext_vector_type(4))) int i32x4;

__device__ __forceinline__ u16 f2bf(float f) {
  unsigned u = __float_as_uint(f);
  u += 0x7FFFu + ((u >> 16) & 1u);   // round-to-nearest-even
  return (u16)(u >> 16);
}

// ---------------- conversion kernels ----------------
__global__ void cvt_bf16_kernel(const float* __restrict__ src, u16* __restrict__ dst, int n) {
  for (int i = blockIdx.x * blockDim.x + threadIdx.x; i < n; i += gridDim.x * blockDim.x)
    dst[i] = f2bf(src[i]);
}

__global__ void cvt_bf16_2d_kernel(const float* __restrict__ src, int spitch,
                                   u16* __restrict__ dst, int dpitch, int rows, int cols) {
  int total = rows * cols;
  for (int i = blockIdx.x * blockDim.x + threadIdx.x; i < total; i += gridDim.x * blockDim.x) {
    int r = i / cols, c = i - r * cols;
    dst[(size_t)r * dpitch + c] = f2bf(src[(size_t)r * spitch + c]);
  }
}

// ---------------- bf16 WMMA GEMM: C = A(MxK) * B(KxN) + bias, epilogue EP ----------------
// EP: 0 = none, 1 = softplus, 2 = residual add
#define BM 128
#define BN 64
#define BK 32
#define APITCH 40   // 80 bytes/row: 16B aligned, dodges bank conflicts (TDM pads to this pitch)

template <int EP>
__global__ __launch_bounds__(256) void gemm_bf16_kernel(
    const u16* __restrict__ A, int lda,
    const u16* __restrict__ B, int ldb,
    const float* __restrict__ bias,
    const float* __restrict__ resid,
    float* __restrict__ C, int ldc,
    int M, int N, int K) {
  __shared__ u16 As[BM][APITCH];    // A tile, filled by TDM (padded pitch)
  __shared__ u16 BsT[BN][APITCH];   // B tile TRANSPOSED: BsT[n][k]

  const int tid  = threadIdx.x;
  const int lane = tid & 31;
  const int wave = tid >> 5;         // 8 waves: 4 (M) x 2 (N)
  const int wm   = wave >> 1;
  const int wn   = wave & 1;
  const int lrow = lane & 15;
  const int lhi  = lane >> 4;

  const int m0 = blockIdx.y * BM;
  const int n0 = blockIdx.x * BN;

  v8f zero = {0.f,0.f,0.f,0.f,0.f,0.f,0.f,0.f};
  v8f acc00 = zero, acc01 = zero, acc10 = zero, acc11 = zero;

  // B tile: 32(K) x 64(N) -> one 8-elem chunk per thread, stored transposed
  const int brow = tid >> 3;             // K row within tile
  const int bcc  = (tid & 7) * 8;        // N col base within tile

  for (int k0 = 0; k0 < K; k0 += BK) {
#if __has_builtin(__builtin_amdgcn_tensor_load_to_lds)
    // --- Tensor Data Mover: DMA A tile (BM x BK bf16, strided) into padded LDS ---
    if (tid < 32) {
      unsigned long long ga =
          (unsigned long long)(uintptr_t)(A + (size_t)m0 * lda + k0);
      u32x4 g0;
      g0[0] = 1u;                                        // count = 1 (valid D#)
      g0[1] = (unsigned)(uintptr_t)(&As[0][0]);          // lds_addr
      g0[2] = (unsigned)(ga & 0xFFFFFFFFull);            // global_addr[31:0]
      g0[3] = (unsigned)((ga >> 32) & 0x01FFFFFFull) | (2u << 30);  // addr[56:32] | type=2
      i32x8 g1;
      g1[0] = (1 << 16)            // data_size = 2 bytes
            | (1 << 20)            // pad_enable
            | (3 << 22)            // pad_interval: every 16 dwords (one 32-elem row)
            | (3 << 25);           // pad_amount: 4 dwords (8 bf16) -> pitch 40
      g1[1] = (int)(((unsigned)K & 0xFFFFu) << 16);                      // tensor_dim0 lo
      g1[2] = (int)(((unsigned)K >> 16) | (((unsigned)M & 0xFFFFu) << 16)); // dim0 hi | dim1 lo
      g1[3] = (int)(((unsigned)M >> 16) | ((unsigned)BK << 16));         // dim1 hi | tile_dim0
      g1[4] = BM;                  // tile_dim1 = 128 rows (tile_dim2 = 0)
      g1[5] = lda;                 // tensor_dim0_stride[31:0]
      g1[6] = 0;                   // stride hi | tensor_dim1_stride lo
      g1[7] = 0;
      i32x4 z4 = {0, 0, 0, 0};     // groups 2/3 unused (2D tensor)
      i32x8 z8 = {0, 0, 0, 0, 0, 0, 0, 0};
      __builtin_amdgcn_tensor_load_to_lds(g0, g1, z4, z4, z8, 0);
    }
#else
    // fallback: manual A tile load (2 chunks of 8 bf16 per thread)
    {
      const int arow0 = tid >> 2;         const int ac0 = (tid & 3) * 8;
      const int arow1 = (tid + 256) >> 2; const int ac1 = ((tid + 256) & 3) * 8;
      uint4 a0 = *reinterpret_cast<const uint4*>(&A[(size_t)(m0 + arow0) * lda + k0 + ac0]);
      uint4 a1 = *reinterpret_cast<const uint4*>(&A[(size_t)(m0 + arow1) * lda + k0 + ac1]);
      *reinterpret_cast<uint4*>(&As[arow0][ac0]) = a0;
      *reinterpret_cast<uint4*>(&As[arow1][ac1]) = a1;
    }
#endif

    // --- B tile: global row-major load, transposed store into LDS ---
    union { uint4 q; unsigned w[4]; } bu;
    bu.q = make_uint4(0u, 0u, 0u, 0u);
    if (n0 + bcc + 8 <= N)
      bu.q = *reinterpret_cast<const uint4*>(&B[(size_t)(k0 + brow) * ldb + n0 + bcc]);
    #pragma unroll
    for (int j = 0; j < 4; ++j) {
      BsT[bcc + 2 * j][brow]     = (u16)(bu.w[j] & 0xFFFFu);
      BsT[bcc + 2 * j + 1][brow] = (u16)(bu.w[j] >> 16);
    }
    if (k0 + BK < K)   // emits global_prefetch_b8 for next B tile
      __builtin_prefetch(&B[(size_t)(k0 + BK + brow) * ldb + n0 + bcc], 0, 1);

#if __has_builtin(__builtin_amdgcn_tensor_load_to_lds)
    if (tid < 32) __builtin_amdgcn_s_wait_tensorcnt(0);   // A tile DMA complete
#endif
    __syncthreads();

    // A fragments (16-bit A 16x32 layout: lane<16 -> K 0-7 & 16-23; lane>=16 -> K 8-15 & 24-31)
    union { v16bf v; uint4 q[2]; } fa0, fa1, fb0, fb1;
    const int ar = wm * 32;
    fa0.q[0] = *reinterpret_cast<const uint4*>(&As[ar + lrow][lhi * 8]);
    fa0.q[1] = *reinterpret_cast<const uint4*>(&As[ar + lrow][lhi * 8 + 16]);
    fa1.q[0] = *reinterpret_cast<const uint4*>(&As[ar + 16 + lrow][lhi * 8]);
    fa1.q[1] = *reinterpret_cast<const uint4*>(&As[ar + 16 + lrow][lhi * 8 + 16]);

    // B fragments from transposed tile: lane holds column N, elems = K lhi*8 .. +7 and lhi*8+16 .. +23 (same as A)
    const int bc = wn * 32;
    fb0.q[0] = *reinterpret_cast<const uint4*>(&BsT[bc + lrow][lhi * 8]);
    fb0.q[1] = *reinterpret_cast<const uint4*>(&BsT[bc + lrow][lhi * 8 + 16]);
    fb1.q[0] = *reinterpret_cast<const uint4*>(&BsT[bc + 16 + lrow][lhi * 8]);
    fb1.q[1] = *reinterpret_cast<const uint4*>(&BsT[bc + 16 + lrow][lhi * 8 + 16]);

    acc00 = __builtin_amdgcn_wmma_f32_16x16x32_bf16(false, fa0.v, false, fb0.v, (short)0, acc00, false, false);
    acc01 = __builtin_amdgcn_wmma_f32_16x16x32_bf16(false, fa0.v, false, fb1.v, (short)0, acc01, false, false);
    acc10 = __builtin_amdgcn_wmma_f32_16x16x32_bf16(false, fa1.v, false, fb0.v, (short)0, acc10, false, false);
    acc11 = __builtin_amdgcn_wmma_f32_16x16x32_bf16(false, fa1.v, false, fb1.v, (short)0, acc11, false, false);

    __syncthreads();
  }

  // C/D layout: VGPR r -> M = r + 8*lhi ; lane&15 -> N
  v8f accs[2][2] = {{acc00, acc01}, {acc10, acc11}};
  #pragma unroll
  for (int i = 0; i < 2; ++i) {
    #pragma unroll
    for (int j = 0; j < 2; ++j) {
      int gn = n0 + wn * 32 + j * 16 + lrow;
      if (gn >= N) continue;
      int gmb = m0 + wm * 32 + i * 16 + lhi * 8;
      float bval = bias ? bias[gn] : 0.f;
      #pragma unroll
      for (int r = 0; r < 8; ++r) {
        int gm = gmb + r;
        float v = accs[i][j][r] + bval;
        if (EP == 1) v = (v > 20.f) ? v : log1pf(__expf(v));
        if (EP == 2) v += resid[(size_t)gm * ldc + gn];
        C[(size_t)gm * ldc + gn] = v;
      }
    }
  }
}

// ---------------- causal depthwise conv1d + spike ----------------
__global__ void conv_spike_kernel(const float* __restrict__ xres,    // [Mrows, ldx], x_in = cols [0,Di)
                                  const float* __restrict__ Wc,      // [Di,1,4]
                                  const float* __restrict__ bc,
                                  float* __restrict__ spk, u16* __restrict__ spk_bf,
                                  int L, int Di, int ldx) {
  int idx = blockIdx.x * blockDim.x + threadIdx.x;   // over Mrows*Di
  int c   = idx % Di;
  int row = idx / Di;      // row = b*L + t
  int t   = row % L;
  float acc = bc[c];
  #pragma unroll
  for (int j = 0; j < 4; ++j) {
    int tt = t - 3 + j;
    if (tt >= 0) acc += Wc[c * 4 + j] * xres[(size_t)(row - 3 + j) * ldx + c];
  }
  float s = (acc >= 1.0f) ? 1.0f : 0.0f;
  spk[idx]    = s;
  spk_bf[idx] = f2bf(s);
}

// ---------------- selective scan (one lane per (b, channel), 16 states in regs) ----------------
__global__ __launch_bounds__(256) void scan_kernel(
    const float* __restrict__ delta, const float* __restrict__ spk,
    const float* __restrict__ params,            // [Mrows, ldp]; B_p at col Di, C_p at Di+16
    const float* __restrict__ xres,              // res = cols [Di, 2*Di) of [Mrows, ldx]
    const float* __restrict__ A_log, const float* __restrict__ Dp,
    u16* __restrict__ ybf, int L, int Di, int ldp, int ldx) {
  int d = blockIdx.x * blockDim.x + threadIdx.x;   // 0 .. B*Di
  int b = d / Di;
  int c = d % Di;

  float Av[16], h[16];
  #pragma unroll
  for (int s = 0; s < 16; ++s) { Av[s] = -__expf(A_log[c * 16 + s]); h[s] = 0.f; }
  float dparam = Dp[c];

  for (int t = 0; t < L; ++t) {
    size_t row = (size_t)(b * L + t);
    float dl = delta[row * Di + c];
    float sp = spk[row * Di + c];
    const float* bp = params + row * ldp + Di;       // wave-uniform address -> scalar loads
    const float* cp = bp + 16;
    float dB = dl * sp;
    float y  = sp * dparam;
    #pragma unroll
    for (int s = 0; s < 16; ++s) {
      h[s] = __expf(dl * Av[s]) * h[s] + dB * bp[s];
      y += h[s] * cp[s];
    }
    float r = xres[row * ldx + Di + c];
    y *= r / (1.f + __expf(-r));                      // silu gate
    ybf[row * Di + c] = f2bf(y);
  }
}

// ---------------- LayerNorm + spike ----------------
__global__ __launch_bounds__(256) void ln_spike_kernel(const float* __restrict__ o,
    const float* __restrict__ gamma, const float* __restrict__ beta,
    float* __restrict__ out, int D) {
  __shared__ float s1[256], s2[256];
  int row = blockIdx.x, tid = threadIdx.x;
  const float* orow = o + (size_t)row * D;
  float a = 0.f, b2 = 0.f;
  for (int i = tid; i < D; i += 256) { float v = orow[i]; a += v; b2 += v * v; }
  s1[tid] = a; s2[tid] = b2;
  __syncthreads();
  for (int offr = 128; offr > 0; offr >>= 1) {
    if (tid < offr) { s1[tid] += s1[tid + offr]; s2[tid] += s2[tid + offr]; }
    __syncthreads();
  }
  float mu   = s1[0] / D;
  float var  = s2[0] / D - mu * mu;
  float rstd = rsqrtf(var + 1e-5f);
  for (int i = tid; i < D; i += 256) {
    float nv = (orow[i] - mu) * rstd * gamma[i] + beta[i];
    out[(size_t)row * D + i] = (nv >= 1.0f) ? 1.0f : 0.0f;
  }
}

// ---------------- host ----------------
extern "C" void kernel_launch(void* const* d_in, const int* in_sizes, int n_in,
                              void* d_out, int out_size, void* d_ws, size_t ws_size,
                              hipStream_t stream) {
  (void)in_sizes; (void)n_in; (void)out_size; (void)ws_size;
  const int Bb = 2, L = 1024, Dm = 1024, Di = 2048, S = 16;
  const int LD2 = 2 * Di;          // 4096
  const int PN  = Di + 2 * S;      // 2080
  const int Mrows = Bb * L;        // 2048

  const float* x     = (const float*)d_in[0];
  const float* W_in  = (const float*)d_in[1];
  const float* b_in  = (const float*)d_in[2];
  const float* W_c   = (const float*)d_in[3];
  const float* b_c   = (const float*)d_in[4];
  const float* W_x   = (const float*)d_in[5];
  const float* b_x   = (const float*)d_in[6];
  const float* W_dt  = (const float*)d_in[7];
  const float* b_dt  = (const float*)d_in[8];
  const float* A_log = (const float*)d_in[9];
  const float* Dp    = (const float*)d_in[10];
  const float* W_out = (const float*)d_in[11];
  const float* b_out = (const float*)d_in[12];
  const float* gamma = (const float*)d_in[13];
  const float* beta  = (const float*)d_in[14];

  char* ws = (char*)d_ws;
  size_t off = 0;
  auto alloc = [&](size_t bytes) -> char* {
    char* p = ws + off; off += (bytes + 255) & ~(size_t)255; return p;
  };

  u16*   x_bf    = (u16*)  alloc((size_t)Mrows * Dm * 2);
  u16*   Win_bf  = (u16*)  alloc((size_t)Dm * LD2 * 2);
  u16*   Wx_bf   = (u16*)  alloc((size_t)Di * PN * 2);
  u16*   Wdt_bf  = (u16*)  alloc((size_t)Di * Di * 2);
  u16*   Wout_bf = (u16*)  alloc((size_t)Di * Dm * 2);
  float* xres    = (float*)alloc((size_t)Mrows * LD2 * 4);
  float* spk     = (float*)alloc((size_t)Mrows * Di * 4);
  u16*   spk_bf  = (u16*)  alloc((size_t)Mrows * Di * 2);
  float* params  = (float*)alloc((size_t)Mrows * PN * 4);
  u16*   draw_bf = (u16*)  alloc((size_t)Mrows * Di * 2);
  float* delta   = (float*)alloc((size_t)Mrows * Di * 4);
  u16*   y_bf    = (u16*)  alloc((size_t)Mrows * Di * 2);
  float* o       = (float*)alloc((size_t)Mrows * Dm * 4);

  dim3 blk(256);
  cvt_bf16_kernel<<<1024, blk, 0, stream>>>(x,     x_bf,    Mrows * Dm);
  cvt_bf16_kernel<<<1024, blk, 0, stream>>>(W_in,  Win_bf,  Dm * LD2);
  cvt_bf16_kernel<<<1024, blk, 0, stream>>>(W_x,   Wx_bf,   Di * PN);
  cvt_bf16_kernel<<<1024, blk, 0, stream>>>(W_dt,  Wdt_bf,  Di * Di);
  cvt_bf16_kernel<<<1024, blk, 0, stream>>>(W_out, Wout_bf, Di * Dm);

  // 1) x_and_res = x @ W_in + b_in
  gemm_bf16_kernel<0><<<dim3(LD2 / BN, Mrows / BM), blk, 0, stream>>>(
      x_bf, Dm, Win_bf, LD2, b_in, nullptr, xres, LD2, Mrows, LD2, Dm);
  // 2) causal depthwise conv + spike
  conv_spike_kernel<<<(Mrows * Di) / 256, blk, 0, stream>>>(
      xres, W_c, b_c, spk, spk_bf, L, Di, LD2);
  // 3) params = spikes @ W_x + b_x   (N = 2080, boundary-guarded)
  gemm_bf16_kernel<0><<<dim3((PN + BN - 1) / BN, Mrows / BM), blk, 0, stream>>>(
      spk_bf, Di, Wx_bf, PN, b_x, nullptr, params, PN, Mrows, PN, Di);
  // 4) delta = softplus(params[:, :Di] @ W_dt + b_dt)
  cvt_bf16_2d_kernel<<<1024, blk, 0, stream>>>(params, PN, draw_bf, Di, Mrows, Di);
  gemm_bf16_kernel<1><<<dim3(Di / BN, Mrows / BM), blk, 0, stream>>>(
      draw_bf, Di, Wdt_bf, Di, b_dt, nullptr, delta, Di, Mrows, Di, Di);
  // 5) selective scan + D-skip + silu gate -> y (bf16)
  scan_kernel<<<(Bb * Di) / 256, blk, 0, stream>>>(
      delta, spk, params, xres, A_log, Dp, y_bf, L, Di, PN, LD2);
  // 6) o = x + y @ W_out + b_out
  gemm_bf16_kernel<2><<<dim3(Dm / BN, Mrows / BM), blk, 0, stream>>>(
      y_bf, Di, Wout_bf, Dm, b_out, x, o, Dm, Mrows, Dm, Di);
  // 7) layernorm + spike
  ln_spike_kernel<<<Mrows, blk, 0, stream>>>(o, gamma, beta, (float*)d_out, Dm);
}
